// TensorstaxModel_3393024164226
// MI455X (gfx1250) — matrix operand order repaired and checked by reading, not compiled, not run on hardware
//
#include <hip/hip_runtime.h>

// ---------------- problem constants ----------------
#define B_        32
#define WH        512
#define CH        3
#define KS        32               // PSF is 32x32
#define TY        16               // y-tile = WMMA M
#define XW        4                // x rows per block (one per wave32)
#define THREADS   128              // 4 waves
#define STRIP_ROWS (XW + KS - 1)   // 35 image rows per block
#define STRIP_COLS 48              // 47 needed, pad to 48
#define PSF_CHUNK  8               // kernel rows per LDS chunk
#define PSF_BSTRIDE 264            // 256 + 8 pad (bf16) -> TDM pad fields
#define PRED_ELEMS (B_*WH*WH*CH)   // 25,165,824

// ---------------- workspace layout (bf16 planes) ----------------
// Padded planar image: [c][543][544]  (row r == x-15, col cc == y-15, zeros outside)
#define IMG_ROWS   543
#define IMG_COLS   544
#define IMG_PLANE  (IMG_ROWS*IMG_COLS)          // 295,392 elems
#define IMG_TOTAL  (CH*IMG_PLANE)               // 886,176 elems
// Planar PSF: [c][b][dx][dy], 32768 elems / channel
#define PSF_PLANE  (B_*KS*KS)                   // 32,768
#define PSF_TOTAL  (CH*PSF_PLANE)               // 98,304
#define WS_PSF_OFF ((size_t)IMG_TOTAL*2)        // byte offset of psf planes
// required ws bytes: 886176*2 + 98304*2 = 1,968,960

typedef __bf16 v16bf __attribute__((ext_vector_type(16)));
typedef float  v8f   __attribute__((ext_vector_type(8)));
typedef unsigned int u32x4 __attribute__((ext_vector_type(4)));
typedef int          i32x8 __attribute__((ext_vector_type(8)));
typedef int          i32x4 __attribute__((ext_vector_type(4)));

__device__ __forceinline__ __bf16 f2bf(float f) {
  union { float f; unsigned u; } x; x.f = f;
  unsigned r = (x.u + 0x7FFFu + ((x.u >> 16) & 1u)) >> 16;   // RNE
  unsigned short s = (unsigned short)r;
  __bf16 o; __builtin_memcpy(&o, &s, 2);
  return o;
}

__device__ __forceinline__ unsigned lds_off(const void* p) {
  // flat->LDS: low 32 bits of a generic LDS pointer are the wave-relative
  // LDS byte address (ISA 10.2 aperture mapping)
  return (unsigned)(unsigned long long)(uintptr_t)p;
}

// 2D tensor_load_to_lds: tile (tile_d0 x tile_d1) of 2-byte elements from a
// row-major plane with row stride d0_stride (elements).  D# per ISA ch8.
__device__ __forceinline__ void tdm_load_2d(unsigned lds_addr,
                                            unsigned long long gaddr,
                                            unsigned tensor_d0, unsigned tensor_d1,
                                            unsigned d0_stride,
                                            unsigned tile_d0, unsigned tile_d1,
                                            unsigned pad_en, unsigned pad_ivl,
                                            unsigned pad_amt) {
  u32x4 g0;
  g0[0] = 1u;                                            // count=1, user desc
  g0[1] = lds_addr;                                      // [63:32] lds_addr
  g0[2] = (unsigned)(gaddr & 0xFFFFFFFFull);             // global_addr lo
  g0[3] = (unsigned)((gaddr >> 32) & 0x1FFFFFFull) |     // global_addr[56:32]
          (2u << 30);                                    // type=2 ("image")
  i32x8 g1;
  g1[0] = (int)((1u << 16) |                             // data_size=1 -> 2B
                (pad_en << 20) | (pad_ivl << 22) | (pad_amt << 25));
  g1[1] = (int)((tensor_d0 & 0xFFFFu) << 16);            // atomic_addr=0 | td0 lo
  g1[2] = (int)(((tensor_d0 >> 16) & 0xFFFFu) |
                ((tensor_d1 & 0xFFFFu) << 16));
  g1[3] = (int)(((tensor_d1 >> 16) & 0xFFFFu) | (tile_d0 << 16));
  g1[4] = (int)(tile_d1 & 0xFFFFu);                      // tile_dim1 | tile_dim2=0
  g1[5] = (int)d0_stride;                                // td0_stride[31:0]
  g1[6] = 0;                                             // td0_stride[47:32] | td1_stride lo
  g1[7] = 0;
  i32x4 z4 = {0, 0, 0, 0};
#if __clang_major__ >= 23
  i32x8 z8 = {0, 0, 0, 0, 0, 0, 0, 0};
  __builtin_amdgcn_tensor_load_to_lds(g0, g1, z4, z4, z8, 0);
#else
  __builtin_amdgcn_tensor_load_to_lds(g0, g1, z4, z4, 0);
#endif
}

// ---------------- prep kernels (run once per launch) ----------------
__global__ void zero_loss_kernel(float* p) {
  if (threadIdx.x == 0) *p = 0.0f;
}

// est [512,512,3] f32 -> ws padded planar bf16 [c][543][544] (SAME zeros baked in)
__global__ void prep_image_kernel(const float* __restrict__ est,
                                  __bf16* __restrict__ img_pl) {
  int t = blockIdx.x * blockDim.x + threadIdx.x;
  int c = blockIdx.z;
  if (t >= IMG_PLANE) return;
  int r  = t / IMG_COLS;
  int cc = t - r * IMG_COLS;
  int gr = r - 15, gc = cc - 15;
  float v = 0.0f;
  if (gr >= 0 && gr < WH && gc >= 0 && gc < WH)
    v = est[((size_t)gr * WH + gc) * CH + c];
  img_pl[(size_t)c * IMG_PLANE + t] = f2bf(v);
}

// psf [32,32,32,1,3] f32 -> ws planar bf16 [c][b][dx][dy]
__global__ void prep_psf_kernel(const float* __restrict__ psf,
                                __bf16* __restrict__ psf_pl) {
  int t = blockIdx.x * blockDim.x + threadIdx.x;   // == b*1024 + dx*32 + dy
  int c = blockIdx.z;
  if (t >= PSF_PLANE) return;
  psf_pl[(size_t)c * PSF_PLANE + t] = f2bf(psf[(size_t)t * CH + c]);
}

// ---------------- main kernel ----------------
// Per block: channel c, rows x0..x0+3 (one per wave), cols y0..y0+15, all 32
// batches.  GEMM per wave: D[16 px x 32 batch] += A[16 x 1024] * B[1024 x 32]
// as 32 steps of v_wmma_f32_16x16x32_bf16.  All LDS staging done by the
// Tensor Data Mover (wave 0), PSF double-buffered against compute.
__global__ __launch_bounds__(THREADS)
void deconv_wmma_kernel(const float* __restrict__ obs,
                        const __bf16* __restrict__ img_pl,
                        const __bf16* __restrict__ psf_pl,
                        float* __restrict__ out) {
  __shared__ __align__(16) __bf16 s_img[STRIP_ROWS * STRIP_COLS];   // 3360 B
  __shared__ __align__(16) __bf16 s_psf[2][B_ * PSF_BSTRIDE];       // 2x16896 B

  const int tid  = threadIdx.x;
  const int lane = tid & 31;
  const int w    = tid >> 5;
  const int x0   = blockIdx.x * XW;
  const int y0   = blockIdx.y * TY;
  const int c    = blockIdx.z;

  const unsigned long long img_tile = (unsigned long long)(uintptr_t)(
      img_pl + (size_t)c * IMG_PLANE + (size_t)x0 * IMG_COLS + y0);
  const unsigned long long psf_plane = (unsigned long long)(uintptr_t)(
      psf_pl + (size_t)c * PSF_PLANE);

  if (tid < 32) {   // wave 0 drives the TDM (EXEC ignored, per-wave issue)
    // image strip: 48x35 tile, row stride 544 elems, no LDS padding
    tdm_load_2d(lds_off(s_img), img_tile, IMG_COLS, IMG_ROWS, IMG_COLS,
                STRIP_COLS, STRIP_ROWS, 0, 0, 0);
    // psf chunk 0: 256x32 tile, row stride 1024; LDS pad 4 dwords per 128
    // dwords -> 264-bf16 batch stride (bank spread)
    tdm_load_2d(lds_off(&s_psf[0][0]), psf_plane, KS * KS, B_, KS * KS,
                PSF_CHUNK * KS, B_, 1, 6, 3);
  }

  const int j    = lane & 15;   // A row (pixel) / B col (batch)
  const int half = lane >> 4;
  const int x    = x0 + w;
  const int k0   = half * 16;       // A tap base, elements 0..7 (k = 16*half + e)
  const int k1   = half * 16;       // A tap base, elements 8..15 (k = 16*half + e)

  v8f acc0 = {0.f,0.f,0.f,0.f,0.f,0.f,0.f,0.f};
  v8f acc1 = acc0;

  for (int chunk = 0; chunk < KS / PSF_CHUNK; ++chunk) {
    if (tid < 32) __builtin_amdgcn_s_wait_tensorcnt(0);
    __syncthreads();
    if (tid < 32 && chunk + 1 < KS / PSF_CHUNK)   // prefetch next chunk (overlaps compute)
      tdm_load_2d(lds_off(&s_psf[(chunk + 1) & 1][0]),
                  psf_plane + (unsigned long long)(chunk + 1) * (PSF_CHUNK * KS) * 2,
                  KS * KS, B_, KS * KS, PSF_CHUNK * KS, B_, 1, 6, 3);

    const __bf16* sp = &s_psf[chunk & 1][0];
    for (int dxl = 0; dxl < PSF_CHUNK; ++dxl) {
      const int dx = chunk * PSF_CHUNK + dxl;

      // A: 16-bit 16x32 layout; A[j,k] = strip[w+dx][j+k]
      const __bf16* arow = s_img + (w + dx) * STRIP_COLS + j;
      v16bf a;
      #pragma unroll
      for (int e = 0; e < 8; ++e)  a[e] = arow[k0 + e];
      #pragma unroll
      for (int e = 8; e < 16; ++e) a[e] = arow[k1 + e];

      // B: 32x16 layout; element e -> tap dy = half*16 + e; 16B aligned
      union QV { uint4 q[2]; v16bf v; };
      const uint4* bp0 = reinterpret_cast<const uint4*>(
          sp + j * PSF_BSTRIDE + dxl * KS + half * 16);
      const uint4* bp1 = reinterpret_cast<const uint4*>(
          sp + (j + 16) * PSF_BSTRIDE + dxl * KS + half * 16);
      QV u0, u1;
      u0.q[0] = bp0[0]; u0.q[1] = bp0[1];
      u1.q[0] = bp1[0]; u1.q[1] = bp1[1];

      acc0 = __builtin_amdgcn_wmma_f32_16x16x32_bf16(
          false, a, false, u0.v, (short)0, acc0, false, false);
      acc1 = __builtin_amdgcn_wmma_f32_16x16x32_bf16(
          false, a, false, u1.v, (short)0, acc1, false, false);
    }
  }

  // D: vgpr v, lanes 0-15: M=v,N=lane; lanes 16-31: M=v+8,N=lane-16
  const int bb    = j;
  const int mbase = half * 8;
  float lsum = 0.0f;
  #pragma unroll
  for (int v = 0; v < 8; ++v) {
    int y = y0 + mbase + v;
    size_t i0 = (((size_t)bb * WH + x) * WH + y) * CH + c;
    float p0 = acc0[v];
    out[i0] = p0;
    float d0 = obs[i0] - p0;
    lsum += d0 * d0;
    size_t i1 = (((size_t)(bb + 16) * WH + x) * WH + y) * CH + c;
    float p1 = acc1[v];
    out[i1] = p1;
    float d1 = obs[i1] - p1;
    lsum += d1 * d1;
  }
  #pragma unroll
  for (int off = 16; off > 0; off >>= 1)
    lsum += __shfl_down(lsum, off, 32);
  if (lane == 0)
    atomicAdd(out + PRED_ELEMS, lsum * (1.0f / (float)PRED_ELEMS));
}

extern "C" void kernel_launch(void* const* d_in, const int* in_sizes, int n_in,
                              void* d_out, int out_size, void* d_ws, size_t ws_size,
                              hipStream_t stream) {
  (void)in_sizes; (void)n_in; (void)out_size; (void)ws_size;  // needs ~1.97MB ws
  const float* obs = (const float*)d_in[0];   // [32,512,512,3]
  const float* est = (const float*)d_in[1];   // [512,512,3]
  const float* psf = (const float*)d_in[2];   // [32,32,32,1,3]
  float* out = (float*)d_out;                 // predicted flat + loss scalar

  __bf16* img_pl = (__bf16*)d_ws;
  __bf16* psf_pl = (__bf16*)((char*)d_ws + WS_PSF_OFF);

  zero_loss_kernel<<<1, 1, 0, stream>>>(out + PRED_ELEMS);

  dim3 gi((IMG_PLANE + 255) / 256, 1, CH);
  prep_image_kernel<<<gi, 256, 0, stream>>>(est, img_pl);
  dim3 gp((PSF_PLANE + 255) / 256, 1, CH);
  prep_psf_kernel<<<gp, 256, 0, stream>>>(psf, psf_pl);

  dim3 grid(WH / XW, WH / TY, CH);   // 128 x 32 x 3 = 12288 blocks
  deconv_wmma_kernel<<<grid, THREADS, 0, stream>>>(obs, img_pl, psf_pl, out);
}
